// ModernMLP_1073741824594
// MI455X (gfx1250) — compile-verified, matrix operand order checked by reading, not run on hardware
//
#include <hip/hip_runtime.h>
#include <hip/hip_bf16.h>

// ---------------------------------------------------------------------------
// MoE "ModernMLP": gating (softmax/top-2) + 4 Fractal experts (RMSNorm ->
// SwiGLU(D,2D) residual -> LayerScale residual) + 4 plain SwiGLU(D,4D),
// combined with renormalized top-2 weights. GEMMs: bf16 WMMA, fp32 acc,
// 128x64 block tile / 32x32 wave tile, double-buffered LDS filled by
// gfx1250 async global->LDS DMA (ASYNCcnt).
// ---------------------------------------------------------------------------

typedef __bf16 bf16_t;
typedef __attribute__((ext_vector_type(16))) __bf16 v16bf;
typedef __attribute__((ext_vector_type(8)))  float  v8f;

static constexpr int B_TOK = 32768;
static constexpr int DIM   = 1024;
static constexpr int NEXP  = 8;
static constexpr int NFR   = 4;
static constexpr int H_FR  = 2 * DIM;   // 2048
static constexpr int H_SW  = 4 * DIM;   // 4096
static constexpr int KC    = 64;        // K chunk staged per buffer
static constexpr int BM    = 128;       // block M tile
static constexpr int BN    = 64;        // block N tile

// ---------------------------------------------------------------------------
// gfx1250 async global -> LDS copy (ASYNCcnt-tracked DMA, no VGPR staging)
// ---------------------------------------------------------------------------
__device__ __forceinline__ void async_b128(const bf16_t* g, unsigned lds_off) {
    asm volatile("global_load_async_to_lds_b128 %0, %1, off"
                 :: "v"(lds_off), "v"(g) : "memory");
}
__device__ __forceinline__ void wait_async0() {
    asm volatile("s_wait_asynccnt 0x0" ::: "memory");
}
__device__ __forceinline__ unsigned lds_addr(const void* p) {
    // generic shared pointer: low 32 bits are the LDS offset (ISA flat->LDS rule)
    return (unsigned)(unsigned long long)p;
}

// ---------------------------------------------------------------------------
// Elementwise helpers
// ---------------------------------------------------------------------------
__global__ __launch_bounds__(256) void zero_f32_kernel(float* __restrict__ p, int n4) {
    int i = blockIdx.x * 256 + threadIdx.x;
    if (i < n4) {
        float4 z = {0.f, 0.f, 0.f, 0.f};
        *(float4*)(p + (size_t)i * 4) = z;
    }
}

__global__ __launch_bounds__(256) void cvt_f32_bf16_kernel(const float* __restrict__ in,
                                                           bf16_t* __restrict__ out, int n4) {
    int i = blockIdx.x * 256 + threadIdx.x;
    if (i < n4) {
        size_t b = (size_t)i * 4;
        float4 v = *(const float4*)(in + b);
        out[b + 0] = (bf16_t)v.x;
        out[b + 1] = (bf16_t)v.y;
        out[b + 2] = (bf16_t)v.z;
        out[b + 3] = (bf16_t)v.w;
    }
}

// ---------------------------------------------------------------------------
// Gating: one wave32 per token. logits = x @ gate_w.T, softmax, top-2,
// renormalize -> comb[B, 8] (zeros for unselected experts).
// ---------------------------------------------------------------------------
__global__ __launch_bounds__(256) void gate_kernel(const float* __restrict__ x,
                                                   const float* __restrict__ gw,
                                                   float* __restrict__ comb) {
    int tok  = blockIdx.x * 8 + (threadIdx.x >> 5);
    int lane = threadIdx.x & 31;
    if (tok >= B_TOK) return;
    const float* xr = x + (size_t)tok * DIM;
    float l[NEXP];
#pragma unroll
    for (int e = 0; e < NEXP; ++e) l[e] = 0.f;
    for (int i = lane; i < DIM; i += 32) {
        float xv = xr[i];
#pragma unroll
        for (int e = 0; e < NEXP; ++e) l[e] += xv * gw[(size_t)e * DIM + i];
    }
#pragma unroll
    for (int off = 16; off >= 1; off >>= 1) {
#pragma unroll
        for (int e = 0; e < NEXP; ++e) l[e] += __shfl_xor(l[e], off, 32);
    }
    if (lane == 0) {
        float m = l[0];
#pragma unroll
        for (int e = 1; e < NEXP; ++e) m = fmaxf(m, l[e]);
        float p[NEXP];
#pragma unroll
        for (int e = 0; e < NEXP; ++e) p[e] = __expf(l[e] - m);
        int i1 = 0;
#pragma unroll
        for (int e = 1; e < NEXP; ++e) if (p[e] > p[i1]) i1 = e;
        int i2 = (i1 == 0) ? 1 : 0;
#pragma unroll
        for (int e = 0; e < NEXP; ++e) if (e != i1 && p[e] > p[i2]) i2 = e;
        float denom = p[i1] + p[i2];
        float w1 = p[i1] / denom, w2 = p[i2] / denom;
#pragma unroll
        for (int e = 0; e < NEXP; ++e)
            comb[(size_t)tok * NEXP + e] = (e == i1) ? w1 : (e == i2) ? w2 : 0.f;
    }
}

// ---------------------------------------------------------------------------
// Per-token inverse RMS + bf16 copy of x.
// ---------------------------------------------------------------------------
__global__ __launch_bounds__(256) void rms_cvt_kernel(const float* __restrict__ x,
                                                      float* __restrict__ rms,
                                                      bf16_t* __restrict__ xb) {
    int tok  = blockIdx.x * 8 + (threadIdx.x >> 5);
    int lane = threadIdx.x & 31;
    if (tok >= B_TOK) return;
    const float* xr = x + (size_t)tok * DIM;
    bf16_t* xo = xb + (size_t)tok * DIM;
    float s = 0.f;
    for (int i = lane; i < DIM; i += 32) {
        float v = xr[i];
        s += v * v;
        xo[i] = (bf16_t)v;
    }
#pragma unroll
    for (int off = 16; off >= 1; off >>= 1) s += __shfl_xor(s, off, 32);
    if (lane == 0) rms[tok] = rsqrtf(s * (1.f / DIM) + 1e-6f);
}

// ---------------------------------------------------------------------------
// Build bf16 A-matrix for a fractal expert: an = bf16(x * rms * norm_e)
// ---------------------------------------------------------------------------
__global__ __launch_bounds__(256) void build_xn_kernel(const float* __restrict__ x,
                                                       const float* __restrict__ rms,
                                                       const float* __restrict__ normw,
                                                       bf16_t* __restrict__ an) {
    int i = blockIdx.x * 256 + threadIdx.x;
    size_t base = (size_t)i * 4;
    if (base >= (size_t)B_TOK * DIM) return;
    int tok = (int)(base / DIM);
    int d   = (int)(base - (size_t)tok * DIM);
    float r = rms[tok];
    float4 xv = *(const float4*)(x + base);
    an[base + 0] = (bf16_t)(xv.x * r * normw[d + 0]);
    an[base + 1] = (bf16_t)(xv.y * r * normw[d + 1]);
    an[base + 2] = (bf16_t)(xv.z * r * normw[d + 2]);
    an[base + 3] = (bf16_t)(xv.w * r * normw[d + 3]);
}

// ---------------------------------------------------------------------------
// Fused up-projection: ACT = silu(A @ W1^T) * (A @ W3^T), bf16 in/out.
// Block: 256 threads = 8 waves; 128x64 block tile; wave -> 32(M) x 32(N)
// (2x2 WMMA subtiles). Double-buffered LDS (KC=64) via async DMA.
// ---------------------------------------------------------------------------
__global__ __launch_bounds__(256) void gemm1_swiglu_kernel(
    const bf16_t* __restrict__ A,    // [M x K]
    const bf16_t* __restrict__ W1,   // [H x K]
    const bf16_t* __restrict__ W3,   // [H x K]
    bf16_t* __restrict__ ACT,        // [M x H]
    int K, int H) {
    __shared__ __align__(32) bf16_t Als[2][BM * KC];
    __shared__ __align__(32) bf16_t B1s[2][BN * KC];
    __shared__ __align__(32) bf16_t B3s[2][BN * KC];

    const int m0   = blockIdx.x * BM;
    const int n0   = blockIdx.y * BN;
    const int t    = threadIdx.x;
    const int w    = t >> 5;
    const int lane = t & 31;
    const int wm   = (w >> 1) * 32;     // wave M sub-offset (0,32,64,96)
    const int wn   = (w & 1) * 32;      // wave N sub-offset (0,32)
    // A staging: 128 rows x 64 cols, 32 elems (64B) per thread
    const int arow = t >> 1, acol = (t & 1) * 32;
    // B staging: 64 rows x 64 cols, 16 elems (32B) per thread
    const int brow = t >> 2, bcol = (t & 3) * 16;

    const int mfr = wm + (lane & 15);   // A-fragment row (subtile 0; +16 for 1)
    const int nfr = lane & 15;          // B-fragment col within subtile
    const int kfo = (lane >> 4) * 16;   // K half select

    v8f acc1[2][2] = {};                // [ms][ns] for W1 path
    v8f acc3[2][2] = {};                // [ms][ns] for W3 path

    auto stage = [&](int buf, int k0) {
        const bf16_t* ga = &A [(size_t)(m0 + arow) * K + k0 + acol];
        const bf16_t* g1 = &W1[(size_t)(n0 + brow) * K + k0 + bcol];
        const bf16_t* g3 = &W3[(size_t)(n0 + brow) * K + k0 + bcol];
        unsigned la = lds_addr(&Als[buf][arow * KC + acol]);
        unsigned l1 = lds_addr(&B1s[buf][brow * KC + bcol]);
        unsigned l3 = lds_addr(&B3s[buf][brow * KC + bcol]);
        async_b128(ga,      la);      async_b128(ga + 8,  la + 16);
        async_b128(ga + 16, la + 32); async_b128(ga + 24, la + 48);
        async_b128(g1,      l1);      async_b128(g1 + 8,  l1 + 16);
        async_b128(g3,      l3);      async_b128(g3 + 8,  l3 + 16);
    };

    stage(0, 0);
    int buf = 0;
    for (int k0 = 0; k0 < K; k0 += KC, buf ^= 1) {
        wait_async0();        // this wave's DMA into `buf` complete
        __syncthreads();      // everyone's DMA complete; prev buffer fully read
        if (k0 + KC < K) stage(buf ^ 1, k0 + KC);
#pragma unroll
        for (int kk = 0; kk < KC; kk += 32) {
            v16bf af[2], b1f[2], b3f[2];
#pragma unroll
            for (int s = 0; s < 2; ++s) {
                af [s] = *(const v16bf*)&Als[buf][(mfr + s * 16) * KC + kk + kfo];
                b1f[s] = *(const v16bf*)&B1s[buf][(wn + s * 16 + nfr) * KC + kk + kfo];
                b3f[s] = *(const v16bf*)&B3s[buf][(wn + s * 16 + nfr) * KC + kk + kfo];
            }
#pragma unroll
            for (int ms = 0; ms < 2; ++ms)
#pragma unroll
                for (int ns = 0; ns < 2; ++ns) {
                    acc1[ms][ns] = __builtin_amdgcn_wmma_f32_16x16x32_bf16(
                        false, af[ms], false, b1f[ns], (short)0, acc1[ms][ns], false, false);
                    acc3[ms][ns] = __builtin_amdgcn_wmma_f32_16x16x32_bf16(
                        false, af[ms], false, b3f[ns], (short)0, acc3[ms][ns], false, false);
                }
        }
    }

    // Epilogue: act = silu(h1) * h3, cast bf16.
    const int mbase = m0 + wm + (lane >> 4) * 8;   // + ms*16 + r
    const int ncol  = n0 + wn + (lane & 15);       // + ns*16
#pragma unroll
    for (int ms = 0; ms < 2; ++ms)
#pragma unroll
        for (int ns = 0; ns < 2; ++ns)
#pragma unroll
            for (int r = 0; r < 8; ++r) {
                float h1 = acc1[ms][ns][r], h3 = acc3[ms][ns][r];
                float a = (h1 / (1.f + __expf(-h1))) * h3;
                ACT[(size_t)(mbase + ms * 16 + r) * H + ncol + ns * 16] = (bf16_t)a;
            }
}

// ---------------------------------------------------------------------------
// Down-projection + combine epilogue:
//   c = ACT @ W2^T                      [B x D] (fp32 acc)
//   mode 1 (fractal): eo = gamma*( x*rms*norm + c ) + x
//   mode 0 (plain):   eo = c
//   out += comb[:, expert] * eo
// ---------------------------------------------------------------------------
__global__ __launch_bounds__(256) void gemm2_combine_kernel(
    const bf16_t* __restrict__ ACT,   // [M x K]  (K = hidden)
    const bf16_t* __restrict__ W2,    // [D x K]
    const float* __restrict__ x,      // [M x D]
    const float* __restrict__ rms,    // [M]
    const float* __restrict__ normw,  // [D] (fractal) or null
    const float* __restrict__ gammaw, // [D] (fractal) or null
    const float* __restrict__ comb,   // [M x 8]
    float* __restrict__ out,          // [M x D]
    int K, int expert, int mode) {
    __shared__ __align__(32) bf16_t Als[2][BM * KC];
    __shared__ __align__(32) bf16_t B2s[2][BN * KC];

    const int m0   = blockIdx.x * BM;
    const int n0   = blockIdx.y * BN;
    const int t    = threadIdx.x;
    const int w    = t >> 5;
    const int lane = t & 31;
    const int wm   = (w >> 1) * 32;
    const int wn   = (w & 1) * 32;
    const int arow = t >> 1, acol = (t & 1) * 32;
    const int brow = t >> 2, bcol = (t & 3) * 16;

    const int mfr = wm + (lane & 15);
    const int nfr = lane & 15;
    const int kfo = (lane >> 4) * 16;

    v8f acc[2][2] = {};

    auto stage = [&](int buf, int k0) {
        const bf16_t* ga = &ACT[(size_t)(m0 + arow) * K + k0 + acol];
        const bf16_t* g2 = &W2 [(size_t)(n0 + brow) * K + k0 + bcol];
        unsigned la = lds_addr(&Als[buf][arow * KC + acol]);
        unsigned l2 = lds_addr(&B2s[buf][brow * KC + bcol]);
        async_b128(ga,      la);      async_b128(ga + 8,  la + 16);
        async_b128(ga + 16, la + 32); async_b128(ga + 24, la + 48);
        async_b128(g2,      l2);      async_b128(g2 + 8,  l2 + 16);
    };

    stage(0, 0);
    int buf = 0;
    for (int k0 = 0; k0 < K; k0 += KC, buf ^= 1) {
        wait_async0();
        __syncthreads();
        if (k0 + KC < K) stage(buf ^ 1, k0 + KC);
#pragma unroll
        for (int kk = 0; kk < KC; kk += 32) {
            v16bf af[2], bf[2];
#pragma unroll
            for (int s = 0; s < 2; ++s) {
                af[s] = *(const v16bf*)&Als[buf][(mfr + s * 16) * KC + kk + kfo];
                bf[s] = *(const v16bf*)&B2s[buf][(wn + s * 16 + nfr) * KC + kk + kfo];
            }
#pragma unroll
            for (int ms = 0; ms < 2; ++ms)
#pragma unroll
                for (int ns = 0; ns < 2; ++ns)
                    acc[ms][ns] = __builtin_amdgcn_wmma_f32_16x16x32_bf16(
                        false, af[ms], false, bf[ns], (short)0, acc[ms][ns], false, false);
        }
    }

    const int mbase = m0 + wm + (lane >> 4) * 8;
    const int ncol  = n0 + wn + (lane & 15);
#pragma unroll
    for (int ms = 0; ms < 2; ++ms)
#pragma unroll
        for (int r = 0; r < 8; ++r) {
            int m = mbase + ms * 16 + r;
            float cw = comb[(size_t)m * NEXP + expert];
#pragma unroll
            for (int ns = 0; ns < 2; ++ns) {
                int n   = ncol + ns * 16;
                float c = acc[ms][ns][r];
                size_t oi = (size_t)m * DIM + n;
                float eo;
                if (mode) {
                    float xv = x[oi];
                    float xn = xv * rms[m] * normw[n];
                    eo = gammaw[n] * (xn + c) + xv;
                } else {
                    eo = c;
                }
                out[oi] += cw * eo;
            }
        }
}

// ---------------------------------------------------------------------------
// Host launcher
// ---------------------------------------------------------------------------
extern "C" void kernel_launch(void* const* d_in, const int* in_sizes, int n_in,
                              void* d_out, int out_size, void* d_ws, size_t ws_size,
                              hipStream_t stream) {
    (void)in_sizes; (void)n_in; (void)out_size; (void)ws_size;
    const float* x       = (const float*)d_in[0];
    const float* gate_w  = (const float*)d_in[1];
    const float* fr_norm = (const float*)d_in[2];
    const float* fr_gamma= (const float*)d_in[3];
    const float* fr_w1   = (const float*)d_in[4];
    const float* fr_w2   = (const float*)d_in[5];
    const float* fr_w3   = (const float*)d_in[6];
    const float* sw_w1   = (const float*)d_in[7];
    const float* sw_w2   = (const float*)d_in[8];
    const float* sw_w3   = (const float*)d_in[9];
    float* out = (float*)d_out;

    // Workspace layout (bf16 operands L2-friendly; act streams to HBM).
    char* ws = (char*)d_ws;
    size_t off = 0;
    auto alloc = [&](size_t bytes) { size_t o = off; off += (bytes + 255) & ~(size_t)255; return o; };
    const size_t nFR = (size_t)H_FR * DIM;   // per-expert fractal weight elems
    const size_t nSW = (size_t)H_SW * DIM;   // per-expert plain weight elems

    float*  comb = (float*)(ws + alloc((size_t)B_TOK * NEXP * 4));
    float*  rms  = (float*)(ws + alloc((size_t)B_TOK * 4));
    bf16_t* xb   = (bf16_t*)(ws + alloc((size_t)B_TOK * DIM * 2));
    bf16_t* an   = (bf16_t*)(ws + alloc((size_t)B_TOK * DIM * 2));
    bf16_t* act  = (bf16_t*)(ws + alloc((size_t)B_TOK * H_SW * 2));
    bf16_t* frw1 = (bf16_t*)(ws + alloc(NFR * nFR * 2));
    bf16_t* frw2 = (bf16_t*)(ws + alloc(NFR * nFR * 2));
    bf16_t* frw3 = (bf16_t*)(ws + alloc(NFR * nFR * 2));
    bf16_t* sww1 = (bf16_t*)(ws + alloc((NEXP - NFR) * nSW * 2));
    bf16_t* sww2 = (bf16_t*)(ws + alloc((NEXP - NFR) * nSW * 2));
    bf16_t* sww3 = (bf16_t*)(ws + alloc((NEXP - NFR) * nSW * 2));

    // 1) zero output accumulator
    {
        int n4 = (B_TOK * DIM) / 4;
        zero_f32_kernel<<<(n4 + 255) / 256, 256, 0, stream>>>(out, n4);
    }
    // 2) weights -> bf16 (once; afterwards GEMM operands are L2-resident)
    auto cvt = [&](const float* src, bf16_t* dst, size_t n) {
        int n4 = (int)(n / 4);
        cvt_f32_bf16_kernel<<<(n4 + 255) / 256, 256, 0, stream>>>(src, dst, n4);
    };
    cvt(fr_w1, frw1, NFR * nFR);
    cvt(fr_w2, frw2, NFR * nFR);
    cvt(fr_w3, frw3, NFR * nFR);
    cvt(sw_w1, sww1, (NEXP - NFR) * nSW);
    cvt(sw_w2, sww2, (NEXP - NFR) * nSW);
    cvt(sw_w3, sww3, (NEXP - NFR) * nSW);

    // 3) gating + 4) rms / bf16 x
    gate_kernel<<<B_TOK / 8, 256, 0, stream>>>(x, gate_w, comb);
    rms_cvt_kernel<<<B_TOK / 8, 256, 0, stream>>>(x, rms, xb);

    // 5) fractal experts
    for (int e = 0; e < NFR; ++e) {
        const float* nw = fr_norm  + (size_t)e * DIM;
        const float* gw = fr_gamma + (size_t)e * DIM;
        {
            int n4 = (B_TOK * DIM) / 4;
            build_xn_kernel<<<(n4 + 255) / 256, 256, 0, stream>>>(x, rms, nw, an);
        }
        dim3 g1(B_TOK / BM, H_FR / BN);
        gemm1_swiglu_kernel<<<g1, 256, 0, stream>>>(an, frw1 + (size_t)e * nFR,
                                                    frw3 + (size_t)e * nFR, act, DIM, H_FR);
        dim3 g2(B_TOK / BM, DIM / BN);
        gemm2_combine_kernel<<<g2, 256, 0, stream>>>(act, frw2 + (size_t)e * nFR,
                                                     x, rms, nw, gw, comb, out,
                                                     H_FR, e, /*mode=*/1);
    }
    // 6) plain SwiGLU experts
    for (int j = 0; j < NEXP - NFR; ++j) {
        dim3 g1(B_TOK / BM, H_SW / BN);
        gemm1_swiglu_kernel<<<g1, 256, 0, stream>>>(xb, sww1 + (size_t)j * nSW,
                                                    sww3 + (size_t)j * nSW, act, DIM, H_SW);
        dim3 g2(B_TOK / BM, DIM / BN);
        gemm2_combine_kernel<<<g2, 256, 0, stream>>>(act, sww2 + (size_t)j * nSW,
                                                     x, rms, nullptr, nullptr, comb, out,
                                                     H_SW, NFR + j, /*mode=*/0);
    }
}
